// DbrxExperts_35957466202270
// MI455X (gfx1250) — matrix operand order repaired and checked by reading, not compiled, not run on hardware
//
#include <hip/hip_runtime.h>
#include <hip/hip_bf16.h>
#include <stdint.h>

// ---------------- problem constants ----------------
#define NTOK    4096
#define DMODEL  2048
#define NEXP    16
#define FFN     4096
#define TOPK    4
#define MAXSLOT 4096   // worst case tokens per expert

typedef __attribute__((ext_vector_type(16))) __bf16 v16bf;
typedef __attribute__((ext_vector_type(8)))  float  v8f;

union Frag { v16bf v; uint4 q[2]; };          // 32B WMMA operand
union Pk4  { __bf16 b[4]; uint2 u; };          // 4x f32->bf16 pack

// ---------------- workspace layout (bytes) ----------------
// Xbf      : 4096*2048*2          = 16,777,216
// counts   : 16*4                  (@ 16,777,216)
// offsets  : 17*4                  (@ 16,777,472)
// toklist  : 16*4096*4            = 262,144 (@ 16,777,728)
// pairs    : 4096*4*4             = 65,536  (@ 17,039,872)
// gates    : 4096*4*4             = 65,536  (@ 17,105,408)
// hbuf     : 16384*4096*2 (bf16)  = 134,217,728 (@ 17,170,944)
// pair_out : 16384*2048*4 (f32)   = 134,217,728 (@ 151,388,672)
#define OFF_XBF   0ull
#define OFF_CNT   16777216ull
#define OFF_OFFS  16777472ull
#define OFF_TOK   16777728ull
#define OFF_PAIR  17039872ull
#define OFF_GATE  17105408ull
#define OFF_HBUF  17170944ull
#define OFF_POUT  151388672ull

// =========================================================
// 0) zero the per-expert counters
// =========================================================
__global__ void k_init(int* __restrict__ counts) {
    if (threadIdx.x < NEXP) counts[threadIdx.x] = 0;
}

// =========================================================
// 1) fp32 -> bf16 copy of hidden states (done once, reused 4x per token)
// =========================================================
__global__ __launch_bounds__(256) void k_cvt_x(const float* __restrict__ x,
                                               __bf16* __restrict__ xb) {
    size_t i4 = (size_t)blockIdx.x * blockDim.x + threadIdx.x;   // float4 index
    const float4 v = *(const float4*)(x + i4 * 4);
    Pk4 p;
    p.b[0] = (__bf16)v.x; p.b[1] = (__bf16)v.y;
    p.b[2] = (__bf16)v.z; p.b[3] = (__bf16)v.w;
    *(uint2*)(xb + i4 * 4) = p.u;
}

// =========================================================
// 2) router: one wave32 per token. logits -> softmax -> top4 -> scatter
// =========================================================
__global__ __launch_bounds__(128) void k_router(const float* __restrict__ x,
                                                const float* __restrict__ rw,
                                                int*   __restrict__ counts,
                                                int*   __restrict__ toklist,
                                                int*   __restrict__ pairs,
                                                float* __restrict__ gates) {
    const int wid  = threadIdx.x >> 5;
    const int lane = threadIdx.x & 31;
    const int t    = blockIdx.x * 4 + wid;

    // x row resident in registers: 2048 / 32 lanes = 64 floats per lane
    float xr[64];
    const float* xp = x + (size_t)t * DMODEL;
#pragma unroll
    for (int i = 0; i < 64; ++i) xr[i] = xp[lane + i * 32];

    float lg[NEXP];
#pragma unroll
    for (int e = 0; e < NEXP; ++e) {
        const float* wp = rw + (size_t)e * DMODEL;
        float s = 0.f;
#pragma unroll
        for (int i = 0; i < 64; ++i) s += xr[i] * wp[lane + i * 32];
        // full wave32 xor-reduction -> every lane holds the total
        for (int off = 16; off; off >>= 1) s += __shfl_xor(s, off, 32);
        lg[e] = s;
    }

    if (lane == 0) {
        float m = lg[0];
#pragma unroll
        for (int e = 1; e < NEXP; ++e) m = fmaxf(m, lg[e]);
        float p[NEXP];
#pragma unroll
        for (int e = 0; e < NEXP; ++e) p[e] = __expf(lg[e] - m);

        unsigned used = 0u;
        int   ide[TOPK];
        float val[TOPK];
        float vs = 0.f;
#pragma unroll
        for (int k = 0; k < TOPK; ++k) {
            float bv = -1.f; int bi = 0;
#pragma unroll
            for (int e = 0; e < NEXP; ++e)
                if (!((used >> e) & 1u) && p[e] > bv) { bv = p[e]; bi = e; }
            used |= 1u << bi;
            ide[k] = bi; val[k] = bv; vs += bv;
        }
        const float inv = 1.f / vs;
#pragma unroll
        for (int k = 0; k < TOPK; ++k) {
            const int e  = ide[k];
            const int sl = atomicAdd(&counts[e], 1);
            toklist[e * MAXSLOT + sl] = t;
            pairs[t * TOPK + k]       = (e << 12) | sl;
            gates[t * TOPK + k]       = val[k] * inv;
        }
    }
}

// =========================================================
// 3) exclusive prefix scan of the 16 counts (compacted row offsets)
// =========================================================
__global__ void k_scan(const int* __restrict__ counts, int* __restrict__ offsets) {
    if (threadIdx.x == 0) {
        int a = 0;
        for (int e = 0; e < NEXP; ++e) { offsets[e] = a; a += counts[e]; }
        offsets[NEXP] = a;
    }
}

// =========================================================
// 4) GEMM1 per expert (gathered rows):  h = silu(X@w1^T) * (X@v1^T)
//    block = 256 thr (8 waves), tile 128(tok) x 128(ffn), BK = 32
//    per wave: 32x64 out for BOTH matmuls -> 16 wmma / K-step
// =========================================================
__global__ __launch_bounds__(256) void k_gemm1(const __bf16* __restrict__ Xbf,
                                               const float*  __restrict__ wst,
                                               const int*    __restrict__ counts,
                                               const int*    __restrict__ offsets,
                                               const int*    __restrict__ toklist,
                                               __bf16*       __restrict__ hbuf) {
    const int e     = blockIdx.z;
    const int cnt   = counts[e];
    const int ytile = blockIdx.y;
    if (ytile * 128 >= cnt) return;                 // uniform early-exit
    const int nbase = blockIdx.x * 128;
    const int offs  = offsets[e];

    const int tid  = threadIdx.x;
    const int lane = tid & 31, wid = tid >> 5;
    const int wm = wid & 3, wn = wid >> 2;          // 4 (M) x 2 (N) wave grid
    const int lh = lane >> 4, lm = lane & 15;

    __shared__ __attribute__((aligned(16))) __bf16 As[128 * 40];  // +8 pad vs banks
    __shared__ __attribute__((aligned(16))) __bf16 Bw[128 * 40];
    __shared__ __attribute__((aligned(16))) __bf16 Bv[128 * 40];
    __shared__ int toks[128];

    if (tid < 128) {
        const int g = ytile * 128 + tid;
        toks[tid] = (g < cnt) ? toklist[e * MAXSLOT + g] : -1;
    }
    __syncthreads();

    const int arow = tid >> 1, ahalf = tid & 1;
    const int tokr = toks[arow];

    const float* w1base = wst + (size_t)e * (2 * FFN) * DMODEL;
    const float* v1base = w1base + (size_t)FFN * DMODEL;

    v8f accw[2][4], accv[2][4];
    const v8f vz = {0.f, 0.f, 0.f, 0.f, 0.f, 0.f, 0.f, 0.f};
#pragma unroll
    for (int a = 0; a < 2; ++a)
#pragma unroll
        for (int b = 0; b < 4; ++b) { accw[a][b] = vz; accv[a][b] = vz; }

    for (int kk = 0; kk < DMODEL; kk += 32) {
        // ---- A: 128 gathered token rows x 32 bf16 ----
        {
            uint4 z = {0u, 0u, 0u, 0u}; uint4 d0 = z, d1 = z;
            if (tokr >= 0) {
                const uint4* src = (const uint4*)(Xbf + (size_t)tokr * DMODEL + kk + ahalf * 16);
                d0 = src[0]; d1 = src[1];
            }
            uint4* dst = (uint4*)(As + arow * 40 + ahalf * 16);
            dst[0] = d0; dst[1] = d1;
        }
        // ---- B: w1/v1 tiles 128x32 fp32 -> bf16 into LDS ----
#pragma unroll
        for (int m = 0; m < 4; ++m) {
            const int f = m * 256 + tid;            // float4 index in tile
            const int brow = f >> 3, kq = f & 7;
            const float4 fw = *(const float4*)(w1base + (size_t)(nbase + brow) * DMODEL + kk + kq * 4);
            const float4 fv = *(const float4*)(v1base + (size_t)(nbase + brow) * DMODEL + kk + kq * 4);
            Pk4 pw, pv;
            pw.b[0] = (__bf16)fw.x; pw.b[1] = (__bf16)fw.y; pw.b[2] = (__bf16)fw.z; pw.b[3] = (__bf16)fw.w;
            pv.b[0] = (__bf16)fv.x; pv.b[1] = (__bf16)fv.y; pv.b[2] = (__bf16)fv.z; pv.b[3] = (__bf16)fv.w;
            *(uint2*)(Bw + brow * 40 + kq * 4) = pw.u;
            *(uint2*)(Bv + brow * 40 + kq * 4) = pv.u;
        }
        if (kk + 32 < DMODEL)   // hint the next weight tile toward L2
            __builtin_prefetch(w1base + (size_t)(nbase + arow) * DMODEL + kk + 32, 0, 1);
        __syncthreads();

        // ---- compute: 2(M) x 4(N) tiles, two matmuls ----
        Frag afr[2];
#pragma unroll
        for (int tm = 0; tm < 2; ++tm) {
            const __bf16* p = As + (wm * 32 + tm * 16 + lm) * 40 + lh * 8;  // K 0..7 / 8..15
            afr[tm].q[0] = *(const uint4*)p;
            afr[tm].q[1] = *(const uint4*)(p + 16);                          // K 16..23 / 24..31
        }
#pragma unroll
        for (int tn = 0; tn < 4; ++tn) {
            const __bf16* pw = Bw + (wn * 64 + tn * 16 + lm) * 40 + lh * 8; // K 0..7 / 8..15, then +16
            const __bf16* pv = Bv + (wn * 64 + tn * 16 + lm) * 40 + lh * 8;
            Frag bw, bv;
            bw.q[0] = *(const uint4*)pw; bw.q[1] = *(const uint4*)(pw + 16);
            bv.q[0] = *(const uint4*)pv; bv.q[1] = *(const uint4*)(pv + 16);
#pragma unroll
            for (int tm = 0; tm < 2; ++tm) {
                accw[tm][tn] = __builtin_amdgcn_wmma_f32_16x16x32_bf16(
                    false, afr[tm].v, false, bw.v, (short)0, accw[tm][tn], false, false);
                accv[tm][tn] = __builtin_amdgcn_wmma_f32_16x16x32_bf16(
                    false, afr[tm].v, false, bv.v, (short)0, accv[tm][tn], false, false);
            }
        }
        __syncthreads();
    }

    // ---- fused SiLU-gate epilogue, compacted bf16 store ----
#pragma unroll
    for (int tm = 0; tm < 2; ++tm)
#pragma unroll
        for (int tn = 0; tn < 4; ++tn)
#pragma unroll
            for (int j = 0; j < 8; ++j) {
                const int slot = ytile * 128 + wm * 32 + tm * 16 + j + lh * 8; // C map: M=j / j+8
                if (slot < cnt) {
                    const float aw = accw[tm][tn][j];
                    const float av = accv[tm][tn][j];
                    const float h  = (aw / (1.f + __expf(-aw))) * av;
                    const int col  = nbase + wn * 64 + tn * 16 + lm;
                    hbuf[(size_t)(offs + slot) * FFN + col] = (__bf16)h;
                }
            }
}

// =========================================================
// 5) GEMM2 per expert: pair_out = h @ w2^T  (K = 4096)
// =========================================================
__global__ __launch_bounds__(256) void k_gemm2(const __bf16* __restrict__ hbuf,
                                               const float*  __restrict__ w2s,
                                               const int*    __restrict__ counts,
                                               const int*    __restrict__ offsets,
                                               float*        __restrict__ pout) {
    const int e     = blockIdx.z;
    const int cnt   = counts[e];
    const int ytile = blockIdx.y;
    if (ytile * 128 >= cnt) return;
    const int nbase = blockIdx.x * 128;
    const int offs  = offsets[e];

    const int tid  = threadIdx.x;
    const int lane = tid & 31, wid = tid >> 5;
    const int wm = wid & 3, wn = wid >> 2;
    const int lh = lane >> 4, lm = lane & 15;

    __shared__ __attribute__((aligned(16))) __bf16 As[128 * 40];
    __shared__ __attribute__((aligned(16))) __bf16 Bs[128 * 40];

    const int arow = tid >> 1, ahalf = tid & 1;
    const int aslot = ytile * 128 + arow;
    const float* w2base = w2s + (size_t)e * DMODEL * FFN;

    v8f acc[2][4];
    const v8f vz = {0.f, 0.f, 0.f, 0.f, 0.f, 0.f, 0.f, 0.f};
#pragma unroll
    for (int a = 0; a < 2; ++a)
#pragma unroll
        for (int b = 0; b < 4; ++b) acc[a][b] = vz;

    for (int kk = 0; kk < FFN; kk += 32) {
        {
            uint4 z = {0u, 0u, 0u, 0u}; uint4 d0 = z, d1 = z;
            if (aslot < cnt) {
                const uint4* src = (const uint4*)(hbuf + (size_t)(offs + aslot) * FFN + kk + ahalf * 16);
                d0 = src[0]; d1 = src[1];
            }
            uint4* dst = (uint4*)(As + arow * 40 + ahalf * 16);
            dst[0] = d0; dst[1] = d1;
        }
#pragma unroll
        for (int m = 0; m < 4; ++m) {
            const int f = m * 256 + tid;
            const int brow = f >> 3, kq = f & 7;
            const float4 fw = *(const float4*)(w2base + (size_t)(nbase + brow) * FFN + kk + kq * 4);
            Pk4 pw;
            pw.b[0] = (__bf16)fw.x; pw.b[1] = (__bf16)fw.y; pw.b[2] = (__bf16)fw.z; pw.b[3] = (__bf16)fw.w;
            *(uint2*)(Bs + brow * 40 + kq * 4) = pw.u;
        }
        __syncthreads();

        Frag afr[2];
#pragma unroll
        for (int tm = 0; tm < 2; ++tm) {
            const __bf16* p = As + (wm * 32 + tm * 16 + lm) * 40 + lh * 8;
            afr[tm].q[0] = *(const uint4*)p;
            afr[tm].q[1] = *(const uint4*)(p + 16);
        }
#pragma unroll
        for (int tn = 0; tn < 4; ++tn) {
            const __bf16* pb = Bs + (wn * 64 + tn * 16 + lm) * 40 + lh * 8;
            Frag bf;
            bf.q[0] = *(const uint4*)pb; bf.q[1] = *(const uint4*)(pb + 16);
#pragma unroll
            for (int tm = 0; tm < 2; ++tm)
                acc[tm][tn] = __builtin_amdgcn_wmma_f32_16x16x32_bf16(
                    false, afr[tm].v, false, bf.v, (short)0, acc[tm][tn], false, false);
        }
        __syncthreads();
    }

#pragma unroll
    for (int tm = 0; tm < 2; ++tm)
#pragma unroll
        for (int tn = 0; tn < 4; ++tn)
#pragma unroll
            for (int j = 0; j < 8; ++j) {
                const int slot = ytile * 128 + wm * 32 + tm * 16 + j + lh * 8;
                if (slot < cnt) {
                    const int col = nbase + wn * 64 + tn * 16 + lm;
                    pout[(size_t)(offs + slot) * DMODEL + col] = acc[tm][tn][j];
                }
            }
}

// =========================================================
// 6) gather:  out[t] = sum_k gate[t][k] * pair_out[row(t,k)]
//    fixed summation order (k = 0..3) -> deterministic fp32 result
// =========================================================
__global__ __launch_bounds__(256) void k_gather(const float* __restrict__ pout,
                                                const int*   __restrict__ offsets,
                                                const int*   __restrict__ pairs,
                                                const float* __restrict__ gates,
                                                float*       __restrict__ out) {
    const int idx4 = blockIdx.x * blockDim.x + threadIdx.x;     // float4 index
    const int t    = idx4 >> 9;                                  // 512 float4 / row
    const int col  = (idx4 & 511) * 4;
    float4 acc = {0.f, 0.f, 0.f, 0.f};
#pragma unroll
    for (int k = 0; k < TOPK; ++k) {
        const int pr   = pairs[t * TOPK + k];
        const int e    = pr >> 12;
        const int slot = pr & 4095;
        const int row  = offsets[e] + slot;
        const float g  = gates[t * TOPK + k];
        const float4 v = *(const float4*)(pout + (size_t)row * DMODEL + col);
        acc.x += g * v.x; acc.y += g * v.y; acc.z += g * v.z; acc.w += g * v.w;
    }
    *(float4*)(out + (size_t)t * DMODEL + col) = acc;
}

// =========================================================
extern "C" void kernel_launch(void* const* d_in, const int* in_sizes, int n_in,
                              void* d_out, int out_size, void* d_ws, size_t ws_size,
                              hipStream_t stream) {
    const float* x   = (const float*)d_in[0];   // (4096, 2048)
    const float* rw  = (const float*)d_in[1];   // (16, 2048)
    const float* wst = (const float*)d_in[2];   // (16, 8192, 2048)
    const float* w2s = (const float*)d_in[3];   // (16, 2048, 4096)
    float* out = (float*)d_out;

    uint8_t* W = (uint8_t*)d_ws;
    __bf16* Xbf    = (__bf16*)(W + OFF_XBF);
    int*    counts = (int*)   (W + OFF_CNT);
    int*    offs   = (int*)   (W + OFF_OFFS);
    int*    toklist= (int*)   (W + OFF_TOK);
    int*    pairs  = (int*)   (W + OFF_PAIR);
    float*  gates  = (float*) (W + OFF_GATE);
    __bf16* hbuf   = (__bf16*)(W + OFF_HBUF);
    float*  pout   = (float*) (W + OFF_POUT);

    k_init  <<<1, 32, 0, stream>>>(counts);
    k_cvt_x <<<(NTOK * DMODEL / 4) / 256, 256, 0, stream>>>(x, Xbf);
    k_router<<<NTOK / 4, 128, 0, stream>>>(x, rw, counts, toklist, pairs, gates);
    k_scan  <<<1, 32, 0, stream>>>(counts, offs);

    dim3 g1(FFN / 128, NTOK / 128, NEXP);      // 32 x 32 x 16
    k_gemm1<<<g1, 256, 0, stream>>>(Xbf, wst, counts, offs, toklist, hbuf);

    dim3 g2(DMODEL / 128, NTOK / 128, NEXP);   // 16 x 32 x 16
    k_gemm2<<<g2, 256, 0, stream>>>(hbuf, w2s, counts, offs, pout);

    k_gather<<<(NTOK * DMODEL / 4) / 256, 256, 0, stream>>>(pout, offs, pairs, gates, out);
}
